// EnhancedVectorQuantizer_87840671138289
// MI455X (gfx1250) — matrix operand order repaired and checked by reading, not compiled, not run on hardware
//
#include <hip/hip_runtime.h>

#define N_TOK   32768
#define DIM     256
#define KCODES  1024

typedef __attribute__((ext_vector_type(16))) __bf16 v16bf;
typedef __attribute__((ext_vector_type(8)))  float  v8f;

union FragBF {
    v16bf v;
    uint4 q[2];
};

__device__ __forceinline__ unsigned short f2bf(float f) {
    unsigned int u = __float_as_uint(f);
    u += 0x7FFFu + ((u >> 16) & 1u);   // round-to-nearest-even
    return (unsigned short)(u >> 16);
}

__device__ __forceinline__ unsigned pack2bf(float lo, float hi) {
    return (unsigned)f2bf(lo) | ((unsigned)f2bf(hi) << 16);
}

// ---------------- K1: column partial sums (128 blocks x 256 rows) ----------
__global__ __launch_bounds__(256) void vq_colstats(const float* __restrict__ x,
                                                   float* __restrict__ psum,
                                                   float* __restrict__ psq) {
    const int d = threadIdx.x;
    const int b = blockIdx.x;
    const float* p = x + (size_t)b * 256 * DIM + d;
    float s = 0.f, s2 = 0.f;
    for (int r = 0; r < 256; ++r) {
        float v = p[(size_t)r * DIM];
        s += v; s2 += v * v;
    }
    psum[b * DIM + d] = s;
    psq [b * DIM + d] = s2;
}

// ---------------- K2: finalize BN stats, zero accumulators -----------------
__global__ __launch_bounds__(256) void vq_finalize_stats(
        const float* __restrict__ psum, const float* __restrict__ psq,
        const float* __restrict__ gamma, const float* __restrict__ beta,
        float* __restrict__ scale, float* __restrict__ shift,
        int* __restrict__ counts, float* __restrict__ lat) {
    const int d = threadIdx.x;
    float s = 0.f, s2 = 0.f;
    for (int b = 0; b < 128; ++b) { s += psum[b * DIM + d]; s2 += psq[b * DIM + d]; }
    const float inv_n = 1.0f / (float)N_TOK;
    float mean = s * inv_n;
    float var  = s2 * inv_n - mean * mean;      // biased variance, matches ref
    float sc = gamma[d] * rsqrtf(var + 1e-5f);
    scale[d] = sc;
    shift[d] = beta[d] - mean * sc;
    for (int i = d; i < KCODES; i += 256) counts[i] = 0;
    for (int i = d; i < 2048;   i += 256) lat[i] = 0.f;
}

// ---------------- K3: codebook -> bf16 + e_sq ------------------------------
__global__ __launch_bounds__(256) void vq_prep_codebook(const float* __restrict__ cb,
                                                        unsigned short* __restrict__ cbbf,
                                                        float* __restrict__ esq) {
    const int k = blockIdx.x, d = threadIdx.x;
    float v = cb[k * DIM + d];
    cbbf[k * DIM + d] = f2bf(v);
    __shared__ float red[256];
    red[d] = v * v;
    __syncthreads();
    for (int s = 128; s > 0; s >>= 1) { if (d < s) red[d] += red[d + s]; __syncthreads(); }
    if (d == 0) esq[k] = red[0];
}

// ---------------- K4: WMMA distance GEMM + argmin --------------------------
// Block = 256 threads = 8 waves; each wave owns 16 tokens.  BN normalize +
// fp32->bf16 conversion is fused into the A-fragment load (x read once, no
// intermediate xn buffer).  The 16-code x 256-dim bf16 codebook tile is
// double-buffered through LDS; B fragments are preloaded into registers so
// the 8 WMMAs issue back-to-back.
#define CB_STRIDE 264   // padded ushort stride to avoid LDS bank conflicts
__global__ __launch_bounds__(256) void vq_argmin_wmma(
        const float* __restrict__ x,
        const unsigned short* __restrict__ cbbf,
        const float* __restrict__ esq,
        const float* __restrict__ scale,
        const float* __restrict__ shift,
        int* __restrict__ out_idx) {
    __shared__ float s_esq[KCODES];
    __shared__ float s_sc[DIM], s_sh[DIM];
    __shared__ unsigned short s_cb[2][16 * CB_STRIDE];

    s_sc[threadIdx.x] = scale[threadIdx.x];
    s_sh[threadIdx.x] = shift[threadIdx.x];
    for (int i = threadIdx.x; i < KCODES; i += 256) s_esq[i] = esq[i];

    const int lane    = threadIdx.x & 31;
    const int wave    = threadIdx.x >> 5;
    const int halfsel = (lane >> 4) & 1;
    const int l15     = lane & 15;
    const int tok_base = blockIdx.x * 128 + wave * 16;

    // per-thread staging coordinates: 2 uint4 chunks of the 8 KB tile
    const int c0   = threadIdx.x * 2;
    const int row0 = c0 >> 5,        off0 = (c0 & 31) * 8;
    const int row1 = (c0 + 1) >> 5,  off1 = ((c0 + 1) & 31) * 8;
    unsigned short* d0 = &s_cb[0][0] + row0 * CB_STRIDE + off0;
    unsigned short* d1 = &s_cb[0][0] + row1 * CB_STRIDE + off1;
    const unsigned short* g0 = cbbf + row0 * DIM + off0;   // + ct*16*DIM
    const unsigned short* g1 = cbbf + row1 * DIM + off1;

    __syncthreads();   // s_sc/s_sh ready before A conversion

    // A fragments: normalize + convert on the fly.
    // A 16x32 layout (matches B): lanes 0-15 (row=l15): K 0..15;
    // lanes 16-31: K 16..31.
    FragBF A[8];
    {
        const float* rp = x + (size_t)(tok_base + l15) * DIM;
#pragma unroll
        for (int kk = 0; kk < 8; ++kk) {
#pragma unroll
            for (int h = 0; h < 2; ++h) {
                const int k0 = kk * 32 + halfsel * 16 + h * 8;
                float4 a = *(const float4*)(rp + k0);
                float4 b = *(const float4*)(rp + k0 + 4);
                uint4 p;
                p.x = pack2bf(a.x * s_sc[k0+0] + s_sh[k0+0],
                              a.y * s_sc[k0+1] + s_sh[k0+1]);
                p.y = pack2bf(a.z * s_sc[k0+2] + s_sh[k0+2],
                              a.w * s_sc[k0+3] + s_sh[k0+3]);
                p.z = pack2bf(b.x * s_sc[k0+4] + s_sh[k0+4],
                              b.y * s_sc[k0+5] + s_sh[k0+5]);
                p.w = pack2bf(b.z * s_sc[k0+6] + s_sh[k0+6],
                              b.w * s_sc[k0+7] + s_sh[k0+7]);
                A[kk].q[h] = p;
            }
        }
    }

    float minv[8];
    int   mini[8];
#pragma unroll
    for (int v = 0; v < 8; ++v) { minv[v] = 3.4e38f; mini[v] = 0; }

    // stage tile 0 into buffer 0
    *(uint4*)d0 = *(const uint4*)g0;
    *(uint4*)d1 = *(const uint4*)g1;
    __syncthreads();

    const size_t TILE_G = (size_t)16 * DIM;           // ushorts per tile
    const size_t TILE_L = (size_t)16 * CB_STRIDE;     // ushorts per LDS buffer

    for (int ct = 0; ct < KCODES / 16; ++ct) {
        const int cur = ct & 1;
        const bool have_next = (ct + 1) < KCODES / 16;

        // fetch next tile global -> registers (overlapped with compute)
        uint4 n0, n1;
        if (have_next) {
            n0 = *(const uint4*)(g0 + (size_t)(ct + 1) * TILE_G);
            n1 = *(const uint4*)(g1 + (size_t)(ct + 1) * TILE_G);
        }
        if (ct + 2 < KCODES / 16)
            __builtin_prefetch(g0 + (size_t)(ct + 2) * TILE_G, 0, 0);

        // preload all 8 B fragments so the WMMA chain issues back-to-back.
        // B 32x16 layout used here (same k order as A): lanes 0-15 (col=l15): K 0..15;
        // lanes 16-31: K 16..31 -> 32 contiguous bytes per lane.
        FragBF B[8];
        {
            const unsigned short* bp = &s_cb[0][0] + cur * TILE_L
                                     + l15 * CB_STRIDE + halfsel * 16;
#pragma unroll
            for (int kk = 0; kk < 8; ++kk) {
                B[kk].q[0] = *(const uint4*)(bp + kk * 32);
                B[kk].q[1] = *(const uint4*)(bp + kk * 32 + 8);
            }
        }

        v8f c = {};
#pragma unroll
        for (int kk = 0; kk < 8; ++kk)
            c = __builtin_amdgcn_wmma_f32_16x16x32_bf16(
                    false, A[kk].v, false, B[kk].v, (short)0, c, false, false);

        const int col = ct * 16 + l15;
        const float es = s_esq[col];
#pragma unroll
        for (int v = 0; v < 8; ++v) {
            float score = es - 2.0f * c[v];   // ||e||^2 - 2 x.e  (x_sq constant)
            if (score < minv[v]) { minv[v] = score; mini[v] = col; }
        }

        if (have_next) {
            unsigned short* nb = &s_cb[0][0] + (1 - cur) * TILE_L;
            *(uint4*)(nb + row0 * CB_STRIDE + off0) = n0;
            *(uint4*)(nb + row1 * CB_STRIDE + off1) = n1;
        }
        __syncthreads();
    }

    // reduce over the 16 columns held across lanes (xor within each half-wave)
#pragma unroll
    for (int m = 1; m <= 8; m <<= 1) {
#pragma unroll
        for (int v = 0; v < 8; ++v) {
            float ov = __shfl_xor(minv[v], m, 32);
            int   oi = __shfl_xor(mini[v], m, 32);
            if (ov < minv[v] || (ov == minv[v] && oi < mini[v])) {
                minv[v] = ov; mini[v] = oi;
            }
        }
    }
    if (l15 == 0) {
        int rbase = tok_base + halfsel * 8;   // C layout: VGPR v -> row halfsel*8+v
#pragma unroll
        for (int v = 0; v < 8; ++v) out_idx[rbase + v] = mini[v];
    }
}

// ---------------- K5: gather, straight-through out, latent + histogram -----
__global__ __launch_bounds__(256) void vq_gather(const float* __restrict__ x,
                                                 const float* __restrict__ cb,
                                                 const int* __restrict__ idxs,
                                                 const float* __restrict__ scale,
                                                 const float* __restrict__ shift,
                                                 float* __restrict__ out,
                                                 int* __restrict__ counts,
                                                 float* __restrict__ lat) {
    __shared__ float s_sc[DIM], s_sh[DIM], red[256];
    s_sc[threadIdx.x] = scale[threadIdx.x];
    s_sh[threadIdx.x] = shift[threadIdx.x];
    const int tb = blockIdx.x * 16;
    if (threadIdx.x < 16) atomicAdd(&counts[idxs[tb + threadIdx.x]], 1);
    __syncthreads();

    float acc = 0.f;
    for (int i = threadIdx.x; i < 16 * DIM; i += 256) {
        int tk = tb + (i >> 8);
        int d  = i & (DIM - 1);
        float q  = cb[(size_t)idxs[tk] * DIM + d];
        float xn = x[(size_t)tk * DIM + d] * s_sc[d] + s_sh[d];
        out[(size_t)tk * DIM + d] = xn + (q - xn);   // straight-through
        float diff = q - xn;
        acc += diff * diff;
    }
    red[threadIdx.x] = acc;
    __syncthreads();
    for (int s = 128; s > 0; s >>= 1) {
        if (threadIdx.x < s) red[threadIdx.x] += red[threadIdx.x + s];
        __syncthreads();
    }
    if (threadIdx.x == 0) lat[blockIdx.x] = red[0];
}

// ---------------- K6: scalars ----------------------------------------------
__global__ __launch_bounds__(256) void vq_final(const int* __restrict__ counts,
                                                const float* __restrict__ lat,
                                                float* __restrict__ out_scalars) {
    __shared__ float red[256];
    const int t = threadIdx.x;
    float ent = 0.f;
    for (int i = t; i < KCODES; i += 256) {
        float p = (float)counts[i] * (1.0f / (float)N_TOK);
        ent -= p * logf(p + 1e-10f);
    }
    red[t] = ent; __syncthreads();
    for (int s = 128; s > 0; s >>= 1) { if (t < s) red[t] += red[t + s]; __syncthreads(); }
    float entropy = red[0];
    __syncthreads();
    float ls = 0.f;
    for (int i = t; i < 2048; i += 256) ls += lat[i];
    red[t] = ls; __syncthreads();
    for (int s = 128; s > 0; s >>= 1) { if (t < s) red[t] += red[t + s]; __syncthreads(); }
    if (t == 0) {
        float e_latent = red[0] / ((float)N_TOK * (float)DIM);
        out_scalars[0] = 0.25f * e_latent - 0.1f * entropy;   // loss
        out_scalars[1] = expf(entropy);                        // perplexity
    }
}

// ---------------- launch ----------------------------------------------------
extern "C" void kernel_launch(void* const* d_in, const int* in_sizes, int n_in,
                              void* d_out, int out_size, void* d_ws, size_t ws_size,
                              hipStream_t stream) {
    (void)in_sizes; (void)n_in; (void)out_size; (void)ws_size;
    const float* x     = (const float*)d_in[0];
    const float* cb    = (const float*)d_in[1];
    const float* gamma = (const float*)d_in[2];
    const float* beta  = (const float*)d_in[3];
    float* out = (float*)d_out;

    char* ws = (char*)d_ws;
    constexpr size_t OFF_CBBF  = 0;                                    // K*D bf16
    constexpr size_t OFF_ESQ   = OFF_CBBF + (size_t)KCODES * DIM * 2;  // K f32
    constexpr size_t OFF_PSUM  = OFF_ESQ  + (size_t)KCODES * 4;        // 128*D f32
    constexpr size_t OFF_PSQ   = OFF_PSUM + (size_t)128 * DIM * 4;
    constexpr size_t OFF_SCALE = OFF_PSQ  + (size_t)128 * DIM * 4;
    constexpr size_t OFF_SHIFT = OFF_SCALE + (size_t)DIM * 4;
    constexpr size_t OFF_IDX   = OFF_SHIFT + (size_t)DIM * 4;          // N int
    constexpr size_t OFF_CNT   = OFF_IDX  + (size_t)N_TOK * 4;         // K int
    constexpr size_t OFF_LAT   = OFF_CNT  + (size_t)KCODES * 4;        // 2048 f32

    unsigned short* cbbf = (unsigned short*)(ws + OFF_CBBF);
    float* esq   = (float*)(ws + OFF_ESQ);
    float* psum  = (float*)(ws + OFF_PSUM);
    float* psq   = (float*)(ws + OFF_PSQ);
    float* scale = (float*)(ws + OFF_SCALE);
    float* shift = (float*)(ws + OFF_SHIFT);
    int*   idxs  = (int*)  (ws + OFF_IDX);
    int*   cnts  = (int*)  (ws + OFF_CNT);
    float* lat   = (float*)(ws + OFF_LAT);

    vq_colstats      <<<128, 256, 0, stream>>>(x, psum, psq);
    vq_finalize_stats<<<1,   256, 0, stream>>>(psum, psq, gamma, beta, scale, shift, cnts, lat);
    vq_prep_codebook <<<KCODES, 256, 0, stream>>>(cb, cbbf, esq);
    vq_argmin_wmma   <<<N_TOK / 128, 256, 0, stream>>>(x, cbbf, esq, scale, shift, idxs);
    vq_gather        <<<N_TOK / 16, 256, 0, stream>>>(x, cb, idxs, scale, shift, out, cnts, lat);
    vq_final         <<<1, 256, 0, stream>>>(cnts, lat, out + (size_t)N_TOK * DIM);
}
